// MultiHeaderAttention_15479062135387
// MI455X (gfx1250) — matrix operand order repaired and checked by reading, not compiled, not run on hardware
//
#include <hip/hip_runtime.h>
#include <hip/hip_bf16.h>
#include <cstdint>

#define DM 1024          // d_model
#define NH 16            // heads
#define HD 64            // head dim
#define BB 4             // batch
#define SS 2048          // seq len
#define MROWS (BB * SS)  // 8192 flattened rows

typedef __attribute__((ext_vector_type(16))) _Float16 v16h;
typedef __attribute__((ext_vector_type(8)))  _Float16 v8h;
typedef __attribute__((ext_vector_type(8)))  float    v8f;
typedef __attribute__((ext_vector_type(4)))  float    v4f;

static __device__ __forceinline__ v8f wmma_f16(v16h a, v16h b, v8f c) {
  // D(f32 16x16) = A(f16 16x32) * B(f16 32x16) + C
  return __builtin_amdgcn_wmma_f32_16x16x32_f16(false, a, false, b, (short)0, c,
                                                false, false);
}

// ---- Fragment loaders (layouts per CDNA5 ISA 7.12.2) ----------------------
// A 16x32 f16: lane row = lane&15; halves i -> K = i + (i>=8?8:0) + (hi?8:0)
static __device__ __forceinline__ v16h a_frag_from_f32(const float* __restrict__ base,
                                                       int ld, int m0, int k0, int lane) {
  const float* p = base + (size_t)(m0 + (lane & 15)) * ld + k0 + ((lane & 16) >> 1);
  v4f x0 = *(const v4f*)(p + 0);
  v4f x1 = *(const v4f*)(p + 4);
  v4f x2 = *(const v4f*)(p + 16);
  v4f x3 = *(const v4f*)(p + 20);
  v16h a;
#pragma unroll
  for (int i = 0; i < 4; ++i) {
    a[i]      = (_Float16)x0[i];
    a[4 + i]  = (_Float16)x1[i];
    a[8 + i]  = (_Float16)x2[i];
    a[12 + i] = (_Float16)x3[i];
  }
  return a;
}

static __device__ __forceinline__ v16h a_frag_from_f16(const _Float16* __restrict__ base,
                                                       int ld, int m0, int k0, int lane) {
  const _Float16* p = base + (size_t)(m0 + (lane & 15)) * ld + k0 + ((lane & 16) >> 1);
  v8h lo = *(const v8h*)(p);
  v8h hi = *(const v8h*)(p + 16);
  v16h a;
#pragma unroll
  for (int i = 0; i < 8; ++i) { a[i] = lo[i]; a[8 + i] = hi[i]; }
  return a;
}

// B 32x16 f16: lane col = lane&15; halves i -> K = i + (i>=8?8:0) + (hi?8:0)
// memory row index = (B-matrix column) n0 + lane&15, contiguous along K.
static __device__ __forceinline__ v16h b_frag_from_f16(const _Float16* __restrict__ base,
                                                       size_t rowStride, int n0, int k0,
                                                       int lane) {
  const _Float16* p = base + (size_t)(n0 + (lane & 15)) * rowStride + k0 + ((lane & 16) >> 1);
  v8h x0 = *(const v8h*)(p);
  v8h x1 = *(const v8h*)(p + 16);
  v16h b;
#pragma unroll
  for (int i = 0; i < 8; ++i) { b[i] = x0[i]; b[8 + i] = x1[i]; }
  return b;
}

// ---- f32 -> f16 weight pre-conversion (weights are L2-resident, reused 8k x)
__global__ __launch_bounds__(256) void cvt_kernel(const float* __restrict__ src,
                                                  _Float16* __restrict__ dst) {
  int i = (blockIdx.x * 256 + threadIdx.x) * 8;
  v4f a = *(const v4f*)(src + i);
  v4f b = *(const v4f*)(src + i + 4);
  v8h o;
#pragma unroll
  for (int j = 0; j < 4; ++j) {
    o[j]     = (_Float16)a[j];
    o[4 + j] = (_Float16)b[j];
  }
  *(v8h*)(dst + i) = o;
}

// ---- Projection GEMM: Y = X(f32) * Wh^T(f16) + bias, f16 output -----------
// vT==0: Y[(b*NH+head)*SS + s][h]  (head-major, for Q and K)
// vT==1: Y[(b*NH+head)*HD + h][s]  (transposed, for V -> contiguous PV B-frags)
static __device__ __forceinline__ void proj_store_tile(_Float16* __restrict__ Y,
                                                       const float* __restrict__ bias,
                                                       v8f c, int o, int bidx, int srow,
                                                       int hi8, int vT) {
  float bv  = bias[o];
  int head  = o >> 6;
  int h     = o & 63;
  if (vT) {
    v8h pack;
#pragma unroll
    for (int r = 0; r < 8; ++r) pack[r] = (_Float16)(c[r] + bv);
    *(v8h*)(Y + ((size_t)(bidx * NH + head) * HD + h) * SS + srow + hi8) = pack;
  } else {
#pragma unroll
    for (int r = 0; r < 8; ++r)
      Y[((size_t)(bidx * NH + head) * SS + (srow + hi8 + r)) * HD + h] =
          (_Float16)(c[r] + bv);
  }
}

__global__ __launch_bounds__(256) void proj_gemm_kernel(const float* __restrict__ X,
                                                        const _Float16* __restrict__ Wh,
                                                        const float* __restrict__ bias,
                                                        _Float16* __restrict__ Y,
                                                        int vT) {
  const int lane = threadIdx.x & 31;
  const int wave = threadIdx.x >> 5;
  const int task = blockIdx.x * 8 + wave;     // (8192/32)*16 = 4096 tasks
  const int OT   = DM / 64;
  const int m0   = (task / OT) * 32;          // 32-row tile per wave
  const int o0   = (task % OT) * 64;

  v8f c0[4] = {{}, {}, {}, {}};
  v8f c1[4] = {{}, {}, {}, {}};
  for (int k0 = 0; k0 < DM; k0 += 32) {
    v16h a0 = a_frag_from_f32(X, DM, m0,      k0, lane);
    v16h a1 = a_frag_from_f32(X, DM, m0 + 16, k0, lane);
#pragma unroll
    for (int t = 0; t < 4; ++t) {
      v16h b = b_frag_from_f16(Wh, DM, o0 + 16 * t, k0, lane);
      c0[t] = wmma_f16(a0, b, c0[t]);
      c1[t] = wmma_f16(a1, b, c1[t]);
    }
  }
  const int lo16 = lane & 15;
  const int hi8  = (lane & 16) >> 1;
  const int bidx = m0 / SS;
  const int srow = m0 % SS;
#pragma unroll
  for (int t = 0; t < 4; ++t) {
    proj_store_tile(Y, bias, c0[t], o0 + 16 * t + lo16, bidx, srow,      hi8, vT);
    proj_store_tile(Y, bias, c1[t], o0 + 16 * t + lo16, bidx, srow + 16, hi8, vT);
  }
}

// ---- Async K-tile staging: 32x64 f16 tile (4KB) via ASYNC LDS engine ------
static __device__ __forceinline__ void stage_k_tile(const _Float16* __restrict__ K,
                                                    int kb, _Float16* lbuf, int lane) {
  unsigned lds0 = (unsigned)(uintptr_t)(void*)lbuf;  // low 32 bits = LDS offset
#pragma unroll
  for (int i = 0; i < 8; ++i) {
    int chunk = lane + 32 * i;  // 256 x 16B chunks; 8 per 128B row
    int row   = chunk >> 3;
    int col8  = chunk & 7;
    unsigned long long ga = (unsigned long long)(uintptr_t)(const void*)
        (K + (size_t)(kb + row) * HD + col8 * 8);
    unsigned la = lds0 + (unsigned)(chunk * 16);
    asm volatile("global_load_async_to_lds_b128 %0, %1, off"
                 :: "v"(la), "v"(ga) : "memory");
  }
}

// ---- Flash attention: one 16-query tile per wave --------------------------
__global__ __launch_bounds__(256) void attn_kernel(const _Float16* __restrict__ qp,
                                                   const _Float16* __restrict__ kp,
                                                   const _Float16* __restrict__ vpT,
                                                   _Float16* __restrict__ ctx) {
  __shared__ __attribute__((aligned(16))) _Float16 kbuf[8][2][32 * HD];  // 64KB
  __shared__ __attribute__((aligned(16))) _Float16 pbuf[8][16 * 32];     // 8KB
  const int lane = threadIdx.x & 31;
  const int wave = threadIdx.x >> 5;
  const int task = blockIdx.x * 8 + wave;   // B*NH*(SS/16) = 8192 tasks
  const int QT   = SS / 16;
  const int qt   = task % QT;
  const int head = (task / QT) % NH;
  const int b    = task / (QT * NH);

  const _Float16* Q = qp  + (size_t)(b * NH + head) * SS * HD;  // (s,h)
  const _Float16* K = kp  + (size_t)(b * NH + head) * SS * HD;  // (s,h)
  const _Float16* V = vpT + (size_t)(b * NH + head) * HD * SS;  // (h,s)

  const int lo16 = lane & 15;
  const int hi8  = (lane & 16) >> 1;

  // Q fragments, pre-scaled by 1/sqrt(HD) = 2^-3 (exact in f16)
  v16h qa0 = a_frag_from_f16(Q, HD, qt * 16, 0,  lane);
  v16h qa1 = a_frag_from_f16(Q, HD, qt * 16, 32, lane);
#pragma unroll
  for (int i = 0; i < 16; ++i) {
    qa0[i] = qa0[i] * (_Float16)0.125f;
    qa1[i] = qa1[i] * (_Float16)0.125f;
  }

  v8f o0 = {}, o1 = {}, o2 = {}, o3 = {};
  float mrow[8], lrow[8];
#pragma unroll
  for (int r = 0; r < 8; ++r) { mrow[r] = -1e30f; lrow[r] = 0.0f; }

  _Float16* P = pbuf[wave];

  stage_k_tile(K, 0, kbuf[wave][0], lane);   // prime buffer 0
  int cur = 0;

  for (int kb = 0; kb < SS; kb += 32) {
    const bool has_next = (kb + 32) < SS;
    if (has_next) {
      stage_k_tile(K, kb + 32, kbuf[wave][cur ^ 1], lane);  // overlap next tile
      __builtin_prefetch(V + (size_t)(kb + 32), 0, 1);
      // async loads complete in order: <=8 outstanding => current tile resident
      asm volatile("s_wait_asynccnt 0x8" ::: "memory");
    } else {
      asm volatile("s_wait_asynccnt 0x0" ::: "memory");
    }
    const _Float16* Kt = kbuf[wave][cur];

    // scores S = (Q/8) * K^T : two 16x16 f32 tiles (keys kb..+15, kb+16..+31)
    v8f s0 = {}, s1 = {};
    {
      v16h b00 = b_frag_from_f16(Kt, HD, 0,  0,  lane);
      v16h b01 = b_frag_from_f16(Kt, HD, 0,  32, lane);
      v16h b10 = b_frag_from_f16(Kt, HD, 16, 0,  lane);
      v16h b11 = b_frag_from_f16(Kt, HD, 16, 32, lane);
      s0 = wmma_f16(qa0, b00, s0);
      s0 = wmma_f16(qa1, b01, s0);
      s1 = wmma_f16(qa0, b10, s1);
      s1 = wmma_f16(qa1, b11, s1);
    }
    // online softmax: rows 0-7 live in lanes 0-15, rows 8-15 in lanes 16-31
    float alpha[8], psum[8];
#pragma unroll
    for (int r = 0; r < 8; ++r) {
      float pm = fmaxf(s0[r], s1[r]);
      pm = fmaxf(pm, __shfl_xor(pm, 8, 16));
      pm = fmaxf(pm, __shfl_xor(pm, 4, 16));
      pm = fmaxf(pm, __shfl_xor(pm, 2, 16));
      pm = fmaxf(pm, __shfl_xor(pm, 1, 16));
      float mnew = fmaxf(mrow[r], pm);
      alpha[r]   = __expf(mrow[r] - mnew);
      mrow[r]    = mnew;
      float p0 = __expf(s0[r] - mnew);
      float p1 = __expf(s1[r] - mnew);
      s0[r] = p0;
      s1[r] = p1;
      float ps = p0 + p1;
      ps += __shfl_xor(ps, 8, 16);
      ps += __shfl_xor(ps, 4, 16);
      ps += __shfl_xor(ps, 2, 16);
      ps += __shfl_xor(ps, 1, 16);
      psum[r] = ps;
    }
#pragma unroll
    for (int r = 0; r < 8; ++r) {
      lrow[r] = lrow[r] * alpha[r] + psum[r];
      o0[r] *= alpha[r];
      o1[r] *= alpha[r];
      o2[r] *= alpha[r];
      o3[r] *= alpha[r];
    }
    // P tile (16x32 f16) -> per-wave LDS, re-read as an A fragment
#pragma unroll
    for (int r = 0; r < 8; ++r) {
      int m = hi8 + r;
      P[m * 32 + lo16]      = (_Float16)s0[r];
      P[m * 32 + 16 + lo16] = (_Float16)s1[r];
    }
    v16h pa  = a_frag_from_f16(P, 32, 0, 0, lane);
    v16h vb0 = b_frag_from_f16(V, SS, 0,  kb, lane);
    v16h vb1 = b_frag_from_f16(V, SS, 16, kb, lane);
    v16h vb2 = b_frag_from_f16(V, SS, 32, kb, lane);
    v16h vb3 = b_frag_from_f16(V, SS, 48, kb, lane);
    o0 = wmma_f16(pa, vb0, o0);
    o1 = wmma_f16(pa, vb1, o1);
    o2 = wmma_f16(pa, vb2, o2);
    o3 = wmma_f16(pa, vb3, o3);

    cur ^= 1;
  }
  // normalize and store ctx as f16 row-major (b, s, d)
  float inv[8];
#pragma unroll
  for (int r = 0; r < 8; ++r) inv[r] = 1.0f / lrow[r];
  _Float16* C = ctx + ((size_t)b * SS + qt * 16) * DM + head * HD;
#pragma unroll
  for (int r = 0; r < 8; ++r) {
    int m = hi8 + r;
    C[(size_t)m * DM +  0 + lo16] = (_Float16)(o0[r] * inv[r]);
    C[(size_t)m * DM + 16 + lo16] = (_Float16)(o1[r] * inv[r]);
    C[(size_t)m * DM + 32 + lo16] = (_Float16)(o2[r] * inv[r]);
    C[(size_t)m * DM + 48 + lo16] = (_Float16)(o3[r] * inv[r]);
  }
}

// ---- Output GEMM: out = ctx(f16) * w_o^T(f16) + b_o  (f32 output) ---------
__global__ __launch_bounds__(256) void out_gemm_kernel(const _Float16* __restrict__ Xh,
                                                       const _Float16* __restrict__ Wh,
                                                       const float* __restrict__ bias,
                                                       float* __restrict__ Y) {
  const int lane = threadIdx.x & 31;
  const int wave = threadIdx.x >> 5;
  const int task = blockIdx.x * 8 + wave;
  const int OT   = DM / 64;
  const int m0   = (task / OT) * 32;
  const int o0   = (task % OT) * 64;

  v8f c0[4] = {{}, {}, {}, {}};
  v8f c1[4] = {{}, {}, {}, {}};
  for (int k0 = 0; k0 < DM; k0 += 32) {
    v16h a0 = a_frag_from_f16(Xh, DM, m0,      k0, lane);
    v16h a1 = a_frag_from_f16(Xh, DM, m0 + 16, k0, lane);
#pragma unroll
    for (int t = 0; t < 4; ++t) {
      v16h b = b_frag_from_f16(Wh, DM, o0 + 16 * t, k0, lane);
      c0[t] = wmma_f16(a0, b, c0[t]);
      c1[t] = wmma_f16(a1, b, c1[t]);
    }
  }
  const int lo16 = lane & 15;
  const int hi8  = (lane & 16) >> 1;
#pragma unroll
  for (int t = 0; t < 4; ++t) {
    int o    = o0 + t * 16 + lo16;
    float bv = bias[o];
#pragma unroll
    for (int r = 0; r < 8; ++r) {
      Y[(size_t)(m0 + hi8 + r) * DM + o]      = c0[t][r] + bv;
      Y[(size_t)(m0 + 16 + hi8 + r) * DM + o] = c1[t][r] + bv;
    }
  }
}

extern "C" void kernel_launch(void* const* d_in, const int* in_sizes, int n_in,
                              void* d_out, int out_size, void* d_ws, size_t ws_size,
                              hipStream_t stream) {
  (void)in_sizes; (void)n_in; (void)out_size; (void)ws_size;
  const float* q   = (const float*)d_in[0];
  const float* k   = (const float*)d_in[1];
  const float* v   = (const float*)d_in[2];
  const float* w_q = (const float*)d_in[3];
  const float* b_q = (const float*)d_in[4];
  const float* w_k = (const float*)d_in[5];
  const float* b_k = (const float*)d_in[6];
  const float* w_v = (const float*)d_in[7];
  const float* b_v = (const float*)d_in[8];
  const float* w_o = (const float*)d_in[9];
  const float* b_o = (const float*)d_in[10];
  float* out = (float*)d_out;

  _Float16* qp  = (_Float16*)d_ws;                 // 16 MB each
  _Float16* kp  = qp  + (size_t)MROWS * DM;
  _Float16* vpT = kp  + (size_t)MROWS * DM;
  _Float16* ctx = vpT + (size_t)MROWS * DM;
  _Float16* whq = ctx + (size_t)MROWS * DM;        // 2 MB each (f16 weights)
  _Float16* whk = whq + (size_t)DM * DM;
  _Float16* whv = whk + (size_t)DM * DM;
  _Float16* who = whv + (size_t)DM * DM;

  dim3 blk(256);
  dim3 cvt_grid((DM * DM) / (8 * 256));            // 512 blocks
  dim3 gemm_grid((MROWS / 32) * (DM / 64) / 8);    // 512 blocks (8 waves each)
  dim3 attn_grid(BB * NH * (SS / 16) / 8);         // 1024 blocks

  cvt_kernel<<<cvt_grid, blk, 0, stream>>>(w_q, whq);
  cvt_kernel<<<cvt_grid, blk, 0, stream>>>(w_k, whk);
  cvt_kernel<<<cvt_grid, blk, 0, stream>>>(w_v, whv);
  cvt_kernel<<<cvt_grid, blk, 0, stream>>>(w_o, who);

  proj_gemm_kernel<<<gemm_grid, blk, 0, stream>>>(q, whq, b_q, qp, 0);
  proj_gemm_kernel<<<gemm_grid, blk, 0, stream>>>(k, whk, b_k, kp, 0);
  proj_gemm_kernel<<<gemm_grid, blk, 0, stream>>>(v, whv, b_v, vpT, 1);
  attn_kernel<<<attn_grid, blk, 0, stream>>>(qp, kp, vpT, ctx);
  out_gemm_kernel<<<gemm_grid, blk, 0, stream>>>(ctx, who, b_o, out);
}
